// HTGNN_30829275251167
// MI455X (gfx1250) — compile-verified, matrix operand order checked by reading, not run on hardware
//
#include <hip/hip_runtime.h>
#include <hip/hip_bf16.h>

// ---------------------------------------------------------------------------
// CDNA5 (gfx1250) WMMA helpers: bf16 A/B fragments, f32 accumulators.
// A: LDS row-major bf16 tile (lane = row M in 0..15, hi-half lanes take K+8).
// B: global bf16 weights pre-swizzled into per-fragment contiguous order:
//    fragment (tile_n, kt) occupies 512 consecutive ushorts, lane*16 each.
// Dependent-WMMA RAW hazards (5 slots for bf16) are avoided by running
// independent accumulator chains per wave and summing at the end.
// ---------------------------------------------------------------------------

typedef __attribute__((ext_vector_type(16))) __bf16          bf16x16;
typedef __attribute__((ext_vector_type(8)))  float           f32x8;
typedef __attribute__((ext_vector_type(8)))  unsigned short  u16x8;

union F8u  { f32x8 v; float f[8]; };
union Frag { u16x8 h[2]; bf16x16 b; };

__device__ __forceinline__ unsigned short f2bf_bits(float x) {
  union { float f; unsigned u; } c; c.f = x;
  unsigned r = c.u + 0x7FFFu + ((c.u >> 16) & 1u);   // round-to-nearest-even
  return (unsigned short)(r >> 16);
}

__device__ __forceinline__ bf16x16 lds_load_a(const unsigned short* As, int lda,
                                              int k0, int lane) {
  int m  = lane & 15;
  int kb = k0 + ((lane >> 4) << 3);                  // +0 or +8
  const unsigned short* p = As + m * lda + kb;
  Frag fr;
  fr.h[0] = *(const u16x8*)(p);                      // K = kb .. kb+7
  fr.h[1] = *(const u16x8*)(p + 16);                 // K = kb+16 .. kb+23
  return fr.b;
}

__device__ __forceinline__ bf16x16 gbl_load_b(const unsigned short* Bs, int ktiles,
                                              int tn, int kt, int lane) {
  const unsigned short* p = Bs + ((size_t)(tn * ktiles + kt) << 9) + (lane << 4);
  Frag fr;
  fr.h[0] = *(const u16x8*)(p);
  fr.h[1] = *(const u16x8*)(p + 8);
  return fr.b;
}

#define WMMA_BF16(a, b, c) \
  __builtin_amdgcn_wmma_f32_16x16x32_bf16(false, (a), false, (b), (short)0, (c), false, false)

// ---------------------------------------------------------------------------
// Weight conversion: f32 row-major [K x N] -> bf16 fragment-swizzled layout.
// ---------------------------------------------------------------------------
__global__ void k_f2bf_swz(const float* __restrict__ src,
                           unsigned short* __restrict__ dst, int K, int N) {
  int idx = blockIdx.x * blockDim.x + threadIdx.x;
  if (idx >= K * N) return;
  int k = idx / N, n = idx - k * N;
  int kk   = k & 31;
  int hi   = (kk >> 3) & 1;                          // K in [8,16)|[24,32) -> hi lanes
  int lane = (n & 15) | (hi << 4);
  int i    = ((kk >> 4) << 3) | (kk & 7);
  int tile = (n >> 4) * (K >> 5) + (k >> 5);
  dst[((size_t)tile << 9) + (lane << 4) + i] = f2bf_bits(src[idx]);
}

// ---------------------------------------------------------------------------
// Projection: h = ReLU(BN(x @ W + b)).   16 rows per block, 8 waves -> 128 cols.
// ---------------------------------------------------------------------------
__global__ void k_proj(const float* __restrict__ x, const unsigned short* __restrict__ Wswz,
                       const float* __restrict__ b, const float* __restrict__ g,
                       const float* __restrict__ beta, const float* __restrict__ rm,
                       const float* __restrict__ rv, float* __restrict__ h,
                       int N, int Cin) {
  __shared__ __align__(16) unsigned short As[16 * 64];
  int tid = threadIdx.x, lane = tid & 31, wave = tid >> 5;
  int row0 = blockIdx.x * 16;
  for (int idx = tid; idx < 16 * Cin; idx += 256) {
    int r = idx / Cin, c = idx - r * Cin;
    int row = row0 + r;
    float v = (row < N) ? x[(size_t)row * Cin + c] : 0.f;
    As[r * Cin + c] = f2bf_bits(v);
  }
  __syncthreads();
  int ktiles = Cin >> 5;
  f32x8 a0 = {}, a1 = {};
  if (ktiles == 2) {                                 // two independent chains
    a0 = WMMA_BF16(lds_load_a(As, Cin, 0,  lane), gbl_load_b(Wswz, 2, wave, 0, lane), a0);
    a1 = WMMA_BF16(lds_load_a(As, Cin, 32, lane), gbl_load_b(Wswz, 2, wave, 1, lane), a1);
  } else {
    a0 = WMMA_BF16(lds_load_a(As, Cin, 0, lane), gbl_load_b(Wswz, 1, wave, 0, lane), a0);
  }
  f32x8 acc = a0 + a1;
  int col = wave * 16 + (lane & 15);
  float inv = rsqrtf(rv[col] + 1e-5f);
  F8u u; u.v = acc;
#pragma unroll
  for (int r = 0; r < 8; ++r) {
    int M = r + ((lane >> 4) << 3);
    int row = row0 + M;
    if (row < N) {
      float val = u.f[r] + b[col];
      val = (val - rm[col]) * inv * g[col] + beta[col];
      h[(size_t)row * 128 + col] = fmaxf(val, 0.f);
    }
  }
}

// ---------------------------------------------------------------------------
// Graph scatter-mean: one wave per edge, lane-parallel 128-d atomic adds.
// ---------------------------------------------------------------------------
__global__ void k_scatter(const float* __restrict__ emb, const int* __restrict__ src,
                          const int* __restrict__ dst, int E,
                          float* __restrict__ agg, float* __restrict__ cnt) {
  int w = (int)((blockIdx.x * (unsigned)blockDim.x + threadIdx.x) >> 5);
  int lane = threadIdx.x & 31;
  if (w >= E) return;
  int s = src[w], d = dst[w];
  const float* sp = emb + (size_t)s * 128;
  float*       dp = agg + (size_t)d * 128;
  __builtin_prefetch(sp, 0, 0);                      // global_prefetch_b8
#pragma unroll
  for (int f = lane; f < 128; f += 32) atomicAdd(&dp[f], sp[f]);
  if (lane == 0) atomicAdd(&cnt[d], 1.f);
}

__global__ void k_fill(float* __restrict__ p, float v, size_t n) {
  size_t i = (size_t)blockIdx.x * blockDim.x + threadIdx.x;
  if (i < n) p[i] = v;
}

// dst = (overwrite ? 0 : dst) + scale * agg / max(cnt,1)   (in-place safe)
__global__ void k_norm_acc(float* __restrict__ dst, const float* __restrict__ agg,
                           const float* __restrict__ cnt, size_t nrows,
                           float scale, int overwrite) {
  size_t i = (size_t)blockIdx.x * blockDim.x + threadIdx.x;
  if (i >= (nrows << 7)) return;
  float v = agg[i] / fmaxf(cnt[i >> 7], 1.f);
  dst[i] = (overwrite ? 0.f : dst[i]) + scale * v;
}

// ---------------------------------------------------------------------------
// Temporal branch: one txn node per block (256 thr / 8 waves).
//   seq = txn_seq @ Wtp + btp + [sin|cos](dt*Wt+bt)   (WMMA, fused epilogue)
//   qkv = seq @ Win + b_in                            (WMMA, 24 tiles, shared A)
//   4-head softmax attention over S=10                (VALU, tiny)
//   out = o @ Wout; temporal = mean_S(out) + b_out    (WMMA)
// ---------------------------------------------------------------------------
__global__ void k_temporal(const float* __restrict__ seq_in, const float* __restrict__ dt,
                           const float* __restrict__ Wt, const float* __restrict__ bt,
                           const unsigned short* __restrict__ Wtp, const float* __restrict__ btp,
                           const unsigned short* __restrict__ Win, const float* __restrict__ b_in,
                           const unsigned short* __restrict__ Wout, const float* __restrict__ b_out,
                           float* __restrict__ tout, int S) {
  __shared__ __align__(16) unsigned short s_a[16 * 136];   // txn_seq bf16, later o bf16
  __shared__ __align__(16) unsigned short s_seq[16 * 136]; // seq bf16
  __shared__ __align__(16) float s_qkv[16 * 392];          // qkv f32, later out f32
  __shared__ float s_sc[4 * 16 * 16];                      // attention scores
  __shared__ float s_dt[16];
  int n = blockIdx.x;
  int tid = threadIdx.x, lane = tid & 31, wave = tid >> 5;

  const float* xs = seq_in + (size_t)n * S * 128;
  for (int idx = tid; idx < 16 * 128; idx += 256) {
    int r = idx >> 7, c = idx & 127;
    s_a[r * 136 + c] = f2bf_bits((r < S) ? xs[r * 128 + c] : 0.f);
  }
  if (tid < 16) s_dt[tid] = (tid < S) ? dt[(size_t)n * S + tid] : 0.f;
  __syncthreads();

  { // stage 1: seq = txn_seq @ Wtp (+ btp + tenc)  -- two chains over K
    f32x8 a0 = {}, a1 = {};
#pragma unroll
    for (int kt = 0; kt < 4; kt += 2) {
      a0 = WMMA_BF16(lds_load_a(s_a, 136, kt * 32, lane),
                     gbl_load_b(Wtp, 4, wave, kt, lane), a0);
      a1 = WMMA_BF16(lds_load_a(s_a, 136, (kt + 1) * 32, lane),
                     gbl_load_b(Wtp, 4, wave, kt + 1, lane), a1);
    }
    f32x8 acc = a0 + a1;
    int col = wave * 16 + (lane & 15);
    float wtc = Wt[col & 63], btc = bt[col & 63], bp = btp[col];
    F8u u; u.v = acc;
#pragma unroll
    for (int r = 0; r < 8; ++r) {
      int M = r + ((lane >> 4) << 3);
      float ph = s_dt[M] * wtc + btc;
      float te = (col < 64) ? sinf(ph) : cosf(ph);
      s_seq[M * 136 + col] = f2bf_bits(u.f[r] + bp + te);
    }
  }
  __syncthreads();

  { // stage 2: qkv -- 3 tiles per wave, interleaved chains, A reused 3x
    f32x8 acc[3] = {f32x8{}, f32x8{}, f32x8{}};
#pragma unroll
    for (int kt = 0; kt < 4; ++kt) {
      bf16x16 a = lds_load_a(s_seq, 136, kt * 32, lane);
#pragma unroll
      for (int j = 0; j < 3; ++j)
        acc[j] = WMMA_BF16(a, gbl_load_b(Win, 4, wave + 8 * j, kt, lane), acc[j]);
    }
#pragma unroll
    for (int j = 0; j < 3; ++j) {
      int col = (wave + 8 * j) * 16 + (lane & 15);
      float bi = b_in[col];
      F8u u; u.v = acc[j];
#pragma unroll
      for (int r = 0; r < 8; ++r) {
        int M = r + ((lane >> 4) << 3);
        s_qkv[M * 392 + col] = u.f[r] + bi;
      }
    }
  }
  __syncthreads();

  // stage 3: attention (S=10, DH=32, 4 heads) in VALU
  const float scl = 0.17677669529663687f;            // 1/sqrt(32)
  for (int idx = tid; idx < 4 * S * S; idx += 256) {
    int hh = idx / (S * S), rem = idx - hh * S * S;
    int i = rem / S, j = rem - i * S;
    const float* q = &s_qkv[i * 392 + hh * 32];
    const float* k = &s_qkv[j * 392 + 128 + hh * 32];
    float s = 0.f;
#pragma unroll
    for (int d = 0; d < 32; ++d) s += q[d] * k[d];
    s_sc[(hh * S + i) * S + j] = s * scl;
  }
  __syncthreads();
  for (int row = tid; row < 4 * S; row += 256) {
    float* p = &s_sc[row * S];
    float m = p[0];
    for (int j = 1; j < S; ++j) m = fmaxf(m, p[j]);
    float sum = 0.f;
    for (int j = 0; j < S; ++j) { float e = __expf(p[j] - m); p[j] = e; sum += e; }
    float inv = 1.f / sum;
    for (int j = 0; j < S; ++j) p[j] *= inv;
  }
  __syncthreads();
  for (int idx = tid; idx < 16 * 128; idx += 256) {  // o = att @ v  -> s_a (bf16)
    int i = idx >> 7, c = idx & 127;
    float o = 0.f;
    if (i < S) {
      int hh = c >> 5;
      const float* att = &s_sc[(hh * S + i) * S];
      for (int j = 0; j < S; ++j) o += att[j] * s_qkv[j * 392 + 256 + c];
    }
    s_a[i * 136 + c] = f2bf_bits(o);
  }
  __syncthreads();

  { // stage 4: out = o @ Wout  (f32 into s_qkv region) -- two chains
    f32x8 a0 = {}, a1 = {};
#pragma unroll
    for (int kt = 0; kt < 4; kt += 2) {
      a0 = WMMA_BF16(lds_load_a(s_a, 136, kt * 32, lane),
                     gbl_load_b(Wout, 4, wave, kt, lane), a0);
      a1 = WMMA_BF16(lds_load_a(s_a, 136, (kt + 1) * 32, lane),
                     gbl_load_b(Wout, 4, wave, kt + 1, lane), a1);
    }
    f32x8 acc = a0 + a1;
    int col = wave * 16 + (lane & 15);
    F8u u; u.v = acc;
#pragma unroll
    for (int r = 0; r < 8; ++r) {
      int M = r + ((lane >> 4) << 3);
      s_qkv[M * 392 + col] = u.f[r];
    }
  }
  __syncthreads();
  if (tid < 128) {
    float s = 0.f;
    for (int r = 0; r < S; ++r) s += s_qkv[r * 392 + tid];
    tout[(size_t)n * 128 + tid] = s / (float)S + b_out[tid];
  }
}

// ---------------------------------------------------------------------------
// Fusion head: 16 nodes per block.
//   gate = sigmoid([h,ctx]@Wg+bg); fused = [h, gate*ctx];
//   logit = ReLU(fused@Wh1+bh1)@Wh2 + bh2
// ---------------------------------------------------------------------------
__global__ void k_head(const float* __restrict__ h_txn, const float* __restrict__ gctx,
                       const float* __restrict__ tctx,
                       const unsigned short* __restrict__ Wg, const float* __restrict__ bg,
                       const unsigned short* __restrict__ Wh1, const float* __restrict__ bh1,
                       const float* __restrict__ Wh2, const float* __restrict__ bh2,
                       float* __restrict__ out, int N) {
  __shared__ __align__(16) unsigned short A1[16 * 264];
  __shared__ float s_ctx[16 * 128];
  __shared__ float s_h[16 * 68];
  int tid = threadIdx.x, lane = tid & 31, wave = tid >> 5;
  int row0 = blockIdx.x * 16;

  for (int idx = tid; idx < 16 * 256; idx += 256) {
    int r = idx >> 8, c = idx & 255;
    int row = row0 + r;
    float v = 0.f;
    if (row < N) {
      if (c < 128) v = h_txn[(size_t)row * 128 + c];
      else { v = gctx[(size_t)row * 128 + (c - 128)] + tctx[(size_t)row * 128 + (c - 128)];
             s_ctx[r * 128 + (c - 128)] = v; }
    } else if (c >= 128) s_ctx[r * 128 + (c - 128)] = 0.f;
    A1[r * 264 + c] = f2bf_bits(v);
  }
  __syncthreads();

  // gate GEMM: K=256 over two independent chains
  f32x8 c0 = {}, c1 = {};
#pragma unroll
  for (int kt = 0; kt < 8; kt += 2) {
    c0 = WMMA_BF16(lds_load_a(A1, 264, kt * 32, lane),
                   gbl_load_b(Wg, 8, wave, kt, lane), c0);
    c1 = WMMA_BF16(lds_load_a(A1, 264, (kt + 1) * 32, lane),
                   gbl_load_b(Wg, 8, wave, kt + 1, lane), c1);
  }
  f32x8 acc = c0 + c1;
  int col = wave * 16 + (lane & 15);
  float gv[8];
  {
    F8u u; u.v = acc;
#pragma unroll
    for (int r = 0; r < 8; ++r) {
      int M = r + ((lane >> 4) << 3);
      float gt = 1.f / (1.f + __expf(-(u.f[r] + bg[col])));
      gv[r] = gt * s_ctx[M * 128 + col];
    }
  }
  __syncthreads();                                   // all waves done reading A1
#pragma unroll
  for (int r = 0; r < 8; ++r) {
    int M = r + ((lane >> 4) << 3);
    A1[M * 264 + 128 + col] = f2bf_bits(gv[r]);      // overwrite ctx half with gate*ctx
  }
  __syncthreads();

  // hidden GEMM: K=256, 4 col tiles (waves 0..3), two chains each
  if (wave < 4) {
    f32x8 h0 = {}, h1 = {};
#pragma unroll
    for (int kt = 0; kt < 8; kt += 2) {
      h0 = WMMA_BF16(lds_load_a(A1, 264, kt * 32, lane),
                     gbl_load_b(Wh1, 8, wave, kt, lane), h0);
      h1 = WMMA_BF16(lds_load_a(A1, 264, (kt + 1) * 32, lane),
                     gbl_load_b(Wh1, 8, wave, kt + 1, lane), h1);
    }
    f32x8 acc2 = h0 + h1;
    int c2 = wave * 16 + (lane & 15);
    F8u u; u.v = acc2;
#pragma unroll
    for (int r = 0; r < 8; ++r) {
      int M = r + ((lane >> 4) << 3);
      s_h[M * 68 + c2] = fmaxf(u.f[r] + bh1[c2], 0.f);
    }
  }
  __syncthreads();

  if (tid < 16) {
    int row = row0 + tid;
    if (row < N) {
      float s = bh2[0];
      for (int c = 0; c < 64; ++c) s += s_h[tid * 68 + c] * Wh2[c];
      out[row] = s;
    }
  }
}

// ---------------------------------------------------------------------------
// Orchestration
// ---------------------------------------------------------------------------
extern "C" void kernel_launch(void* const* d_in, const int* in_sizes, int n_in,
                              void* d_out, int out_size, void* d_ws, size_t ws_size,
                              hipStream_t stream) {
  (void)n_in; (void)out_size;
  const int N_TXN  = in_sizes[0] / 64;
  const int N_CARD = in_sizes[1] / 32;
  const int N_MER  = in_sizes[2] / 32;
  const int N_DEV  = in_sizes[3] / 32;
  const int S      = in_sizes[5] / N_TXN;
  const int E      = in_sizes[6];
  const int E_CC   = in_sizes[12];

  char* ws = (char*)d_ws;
  size_t off = 0;
  auto carve = [&](size_t bytes) -> char* {
    char* p = ws + off;
    off += (bytes + 255) & ~(size_t)255;
    return p;
  };
  float* h_txn  = (float*)carve((size_t)N_TXN  * 128 * 4);
  float* h_card = (float*)carve((size_t)N_CARD * 128 * 4);
  float* h_mer  = (float*)carve((size_t)N_MER  * 128 * 4);
  float* h_dev  = (float*)carve((size_t)N_DEV  * 128 * 4);
  float* shared = (float*)carve((size_t)N_CARD * 128 * 4);
  float* gctx   = (float*)carve((size_t)N_TXN  * 128 * 4);
  float* tctx   = (float*)carve((size_t)N_TXN  * 128 * 4);  // relation tmp, then temporal ctx
  float* cnt    = (float*)carve((size_t)N_TXN * 4);
  unsigned short* Wtp_s  = (unsigned short*)carve(128 * 128 * 2);
  unsigned short* Win_s  = (unsigned short*)carve(128 * 384 * 2);
  unsigned short* Wout_s = (unsigned short*)carve(128 * 128 * 2);
  unsigned short* Wg_s   = (unsigned short*)carve(256 * 128 * 2);
  unsigned short* Wh1_s  = (unsigned short*)carve(256 * 64 * 2);
  unsigned short* Wpt_s  = (unsigned short*)carve(64 * 128 * 2);
  unsigned short* Wpc_s  = (unsigned short*)carve(32 * 128 * 2);
  unsigned short* Wpm_s  = (unsigned short*)carve(32 * 128 * 2);
  unsigned short* Wpd_s  = (unsigned short*)carve(32 * 128 * 2);
  if (off > ws_size) return;

  auto cvt = [&](int idx, unsigned short* dst, int K, int Ncol) {
    int total = K * Ncol;
    k_f2bf_swz<<<(total + 255) / 256, 256, 0, stream>>>((const float*)d_in[idx], dst, K, Ncol);
  };
  cvt(14, Wpt_s, 64, 128);  cvt(20, Wpc_s, 32, 128);
  cvt(26, Wpm_s, 32, 128);  cvt(32, Wpd_s, 32, 128);
  cvt(40, Wtp_s, 128, 128); cvt(42, Win_s, 128, 384); cvt(44, Wout_s, 128, 128);
  cvt(46, Wg_s, 256, 128);  cvt(48, Wh1_s, 256, 64);

  auto proj = [&](int base, unsigned short* Wswz, float* h, int N, int Cin, int xi) {
    k_proj<<<(N + 15) / 16, 256, 0, stream>>>(
        (const float*)d_in[xi], Wswz,
        (const float*)d_in[base + 1], (const float*)d_in[base + 2],
        (const float*)d_in[base + 3], (const float*)d_in[base + 4],
        (const float*)d_in[base + 5], h, N, Cin);
  };
  proj(14, Wpt_s, h_txn,  N_TXN,  64, 0);
  proj(20, Wpc_s, h_card, N_CARD, 32, 1);
  proj(26, Wpm_s, h_mer,  N_MER,  32, 2);
  proj(32, Wpd_s, h_dev,  N_DEV,  32, 3);

  auto fill = [&](float* p, float v, size_t n) {
    k_fill<<<(unsigned)((n + 255) / 256), 256, 0, stream>>>(p, v, n);
  };
  auto relation = [&](const float* emb, int si, int di, int Ecnt, float* agg,
                      float* cv, int nrows, float* dst, float scale, int ovw) {
    fill(agg, 0.f, (size_t)nrows * 128);
    fill(cv, 0.f, (size_t)nrows);
    k_scatter<<<(Ecnt + 7) / 8, 256, 0, stream>>>(emb, (const int*)d_in[si],
                                                  (const int*)d_in[di], Ecnt, agg, cv);
    size_t tot = (size_t)nrows * 128;
    k_norm_acc<<<(unsigned)((tot + 255) / 256), 256, 0, stream>>>(dst, agg, cv,
                                                                  (size_t)nrows, scale, ovw);
  };
  relation(h_card, 6, 7, E, tctx, cnt, N_TXN, gctx, 0.25f, 1);        // card->txn
  relation(h_mer,  8, 9, E, tctx, cnt, N_TXN, gctx, 0.25f, 0);        // merchant->txn
  relation(h_dev, 10, 11, E, tctx, cnt, N_TXN, gctx, 0.25f, 0);       // device->txn
  relation(h_card, 12, 13, E_CC, shared, cnt, N_CARD, shared, 1.f, 1);// card<->card
  relation(shared, 6, 7, E, tctx, cnt, N_TXN, gctx, 0.25f, 0);        // two-hop

  // temporal branch (reuse tctx as temporal_ctx output)
  k_temporal<<<N_TXN, 256, 0, stream>>>(
      (const float*)d_in[4], (const float*)d_in[5],
      (const float*)d_in[38], (const float*)d_in[39],
      Wtp_s, (const float*)d_in[41],
      Win_s, (const float*)d_in[43],
      Wout_s, (const float*)d_in[45],
      tctx, S);

  k_head<<<(N_TXN + 15) / 16, 256, 0, stream>>>(
      h_txn, gctx, tctx,
      Wg_s, (const float*)d_in[47],
      Wh1_s, (const float*)d_in[49],
      (const float*)d_in[50], (const float*)d_in[51],
      (float*)d_out, N_TXN);
}
